// Learned2DRelativeSelfAttention_49349174231256
// MI455X (gfx1250) — matrix operand order repaired and checked by reading, not compiled, not run on hardware
//
#include <hip/hip_runtime.h>
#include <hip/hip_bf16.h>

typedef __attribute__((ext_vector_type(16))) _Float16 v16h;
typedef __attribute__((ext_vector_type(8)))  _Float16 v8h;
typedef __attribute__((ext_vector_type(8)))  float    v8f;

#define NUM_HEADS 8
#define HIDDEN    64
#define HALF_D    32
#define MAXPOS    32
#define BB        8
#define M_TOT     1024   // W*H = 32*32
#define ALLH      512    // NUM_HEADS*HIDDEN

// 1/sqrt(64) * log2(e): folds the softmax scale AND the exp->exp2 base change
// into q and the relative biases, so the attention inner loop has no scaling.
#define SCALE_LOG2 (0.125f * 1.4426950408889634f)

// ---------------------------------------------------------------------------
// Kernel 1: q/k projection (+bias precompute) into WMMA-friendly f16 layouts.
//   qh[b][h][m][d] (f16, pre-scaled), kh[b][h][m][d] (f16)
//   vt[b][d][m]    (f16, transposed V = hidden_states)
//   brow[b][h][m][krow], bcol[b][h][m][kcol]  (f32, pre-scaled biases)
// ---------------------------------------------------------------------------
__global__ void __launch_bounds__(256)
proj_kernel(const float* __restrict__ hs,
            const float* __restrict__ row_emb,
            const float* __restrict__ col_emb,
            const float* __restrict__ Wq, const float* __restrict__ bq,
            const float* __restrict__ Wk, const float* __restrict__ bk,
            _Float16* __restrict__ qh, _Float16* __restrict__ kh,
            _Float16* __restrict__ vt,
            float* __restrict__ brow, float* __restrict__ bcol)
{
    __shared__ float hsrow[HIDDEN];
    __shared__ float qrow[ALLH];

    const int b = blockIdx.x / M_TOT;
    const int m = blockIdx.x % M_TOT;
    const int i = m >> 5;          // query row in 32x32 grid
    const int j = m & 31;          // query col
    const int t = threadIdx.x;

    const float* hp = hs + ((size_t)b * M_TOT + m) * HIDDEN;
    if (t < HIDDEN) hsrow[t] = hp[t];
    __syncthreads();

    for (int c = t; c < ALLH; c += 256) {
        float aq = bq[c], ak = bk[c];
        #pragma unroll 8
        for (int d = 0; d < HIDDEN; ++d) {
            const float x = hsrow[d];
            aq += x * Wq[d * ALLH + c];
            ak += x * Wk[d * ALLH + c];
        }
        qrow[c] = aq;
        const int h  = c >> 6;
        const int d0 = c & 63;
        const size_t o = ((size_t)(b * NUM_HEADS + h) * M_TOT + m) * HIDDEN + d0;
        qh[o] = (_Float16)(aq * SCALE_LOG2);   // scale folded into q
        kh[o] = (_Float16)ak;
    }
    if (t < HIDDEN)
        vt[((size_t)b * HIDDEN + t) * M_TOT + m] = (_Float16)hsrow[t];
    __syncthreads();

    // 256 threads -> (h, k) pairs; separable relative bias dot products
    {
        const int h = t >> 5;
        const int k = t & 31;
        const float* re = row_emb + (size_t)(k - i + MAXPOS - 1) * HALF_D;
        const float* ce = col_emb + (size_t)(k - j + MAXPOS - 1) * HALF_D;
        float sr = 0.f, sc = 0.f;
        #pragma unroll 8
        for (int d = 0; d < HALF_D; ++d) {
            sr += qrow[h * HIDDEN + d]          * re[d];
            sc += qrow[h * HIDDEN + HALF_D + d] * ce[d];
        }
        const size_t o = ((size_t)(b * NUM_HEADS + h) * M_TOT + m) * MAXPOS + k;
        brow[o] = sr * SCALE_LOG2;             // scale folded into bias
        bcol[o] = sc * SCALE_LOG2;
    }
}

// ---------------------------------------------------------------------------
// WMMA fragment loaders (16x16x32 f16). Per CDNA5 ISA §7.12.2:
//  A (16x32): lane%16 = M row; halves interleaved: K = 8*(lane/16) + (j<8?j:j+8)
//  B (32x16): lane%16 = N col; halves interleaved: K = 8*(lane/16) + (j<8?j:j+8)
// ---------------------------------------------------------------------------
__device__ __forceinline__ v16h load_a16(const _Float16* __restrict__ row, int lg) {
    const _Float16* p = row + lg * 8;
    const v8h a = *(const v8h*)p;
    const v8h b = *(const v8h*)(p + 16);
    v16h f;
    #pragma unroll
    for (int t = 0; t < 8; ++t) { f[t] = a[t]; f[8 + t] = b[t]; }
    return f;
}

__device__ __forceinline__ v16h load_b16(const _Float16* __restrict__ row, int lg) {
    const _Float16* p = row + lg * 8;
    const v8h a = *(const v8h*)p;
    const v8h b = *(const v8h*)(p + 16);
    v16h f;
    #pragma unroll
    for (int t = 0; t < 8; ++t) { f[t] = a[t]; f[8 + t] = b[t]; }
    return f;
}

// ---------------------------------------------------------------------------
// Kernel 2: fused flash attention. 1 wave = one 16-row query tile of one
// (b, head); 4 waves/block. Online softmax over 16 key-blocks of 64 keys.
// A 64-key block spans exactly two key-grid rows (krow = 2t, 2t+1); the
// column bias repeats between the halves. Bias is pre-loaded into the WMMA
// C operand so S = bias + Q*K^T comes straight out of the matrix pipe.
// ---------------------------------------------------------------------------
__global__ void __launch_bounds__(128, 1)
attn_kernel(const _Float16* __restrict__ qh,
            const _Float16* __restrict__ kh,
            const _Float16* __restrict__ vt,
            const float* __restrict__ brow,
            const float* __restrict__ bcol,
            float* __restrict__ attn)
{
    __shared__ __align__(16) _Float16 plds[4][16 * 64];

    const int lane = threadIdx.x & 31;
    const int wave = threadIdx.x >> 5;
    const int bh   = blockIdx.x >> 4;           // b*8 + h
    const int b    = bh >> 3;
    const int h    = bh & 7;
    const int mbase = (blockIdx.x & 15) * 64 + wave * 16;
    const int lg = lane >> 4;                   // 16-lane half group
    const int ln = lane & 15;

    // Q tile A-fragments: d 0..31 and d 32..63 (q pre-scaled by 0.125*log2e)
    const _Float16* qrowp = qh + ((size_t)bh * M_TOT + (mbase + ln)) * HIDDEN;
    const v16h aq0 = load_a16(qrowp,      lg);
    const v16h aq1 = load_a16(qrowp + 32, lg);

    float mstat[8], lstat[8], bc0[8], bc1[8];
    v8f acc0 = {}, acc1 = {}, acc2 = {}, acc3 = {};
    const float* browb = brow + ((size_t)bh * M_TOT + mbase + 8 * lg) * MAXPOS;
    #pragma unroll
    for (int r = 0; r < 8; ++r) {
        mstat[r] = -INFINITY;
        lstat[r] = 0.f;
        const int mr = mbase + r + 8 * lg;
        const float* bcp = bcol + ((size_t)bh * M_TOT + mr) * MAXPOS;
        bc0[r] = bcp[ln];          // key col = ln       (subtiles 0 & 2)
        bc1[r] = bcp[16 + ln];     // key col = 16 + ln  (subtiles 1 & 3)
    }

    for (int t = 0; t < 16; ++t) {             // 64 keys per iteration
        // ---- seed C fragments with relative bias ----
        v8f c0, c1, c2, c3;
        #pragma unroll
        for (int r = 0; r < 8; ++r) {
            const float2 brv = *(const float2*)(browb + r * MAXPOS + 2 * t);
            c0[r] = brv.x + bc0[r];            // krow 2t,   kcol ln
            c1[r] = brv.x + bc1[r];            // krow 2t,   kcol 16+ln
            c2[r] = brv.y + bc0[r];            // krow 2t+1, kcol ln
            c3[r] = brv.y + bc1[r];            // krow 2t+1, kcol 16+ln
        }

        // ---- S = bias + Q * K^T for 64 keys (four 16-key subtiles) ----
        const _Float16* kp0 = kh + ((size_t)bh * M_TOT + (t * 64 + ln)) * HIDDEN;
        const _Float16* kp1 = kp0 + 16 * HIDDEN;
        const _Float16* kp2 = kp0 + 32 * HIDDEN;
        const _Float16* kp3 = kp0 + 48 * HIDDEN;
        c0 = __builtin_amdgcn_wmma_f32_16x16x32_f16(false, aq0, false, load_b16(kp0,      lg), (short)0, c0, false, false);
        c0 = __builtin_amdgcn_wmma_f32_16x16x32_f16(false, aq1, false, load_b16(kp0 + 32, lg), (short)0, c0, false, false);
        c1 = __builtin_amdgcn_wmma_f32_16x16x32_f16(false, aq0, false, load_b16(kp1,      lg), (short)0, c1, false, false);
        c1 = __builtin_amdgcn_wmma_f32_16x16x32_f16(false, aq1, false, load_b16(kp1 + 32, lg), (short)0, c1, false, false);
        c2 = __builtin_amdgcn_wmma_f32_16x16x32_f16(false, aq0, false, load_b16(kp2,      lg), (short)0, c2, false, false);
        c2 = __builtin_amdgcn_wmma_f32_16x16x32_f16(false, aq1, false, load_b16(kp2 + 32, lg), (short)0, c2, false, false);
        c3 = __builtin_amdgcn_wmma_f32_16x16x32_f16(false, aq0, false, load_b16(kp3,      lg), (short)0, c3, false, false);
        c3 = __builtin_amdgcn_wmma_f32_16x16x32_f16(false, aq1, false, load_b16(kp3 + 32, lg), (short)0, c3, false, false);

        __syncthreads();    // plds reusable (previous iteration fully consumed)

        // ---- online softmax in base-2 domain ----
        #pragma unroll
        for (int r = 0; r < 8; ++r) {
            const float s0 = c0[r], s1 = c1[r], s2 = c2[r], s3 = c3[r];
            float bm = fmaxf(fmaxf(s0, s1), fmaxf(s2, s3));
            #pragma unroll
            for (int off = 1; off < 16; off <<= 1)
                bm = fmaxf(bm, __shfl_xor(bm, off, 32));
            const float mn = fmaxf(mstat[r], bm);

            const float p0 = exp2f(s0 - mn);
            const float p1 = exp2f(s1 - mn);
            const float p2 = exp2f(s2 - mn);
            const float p3 = exp2f(s3 - mn);
            float rs = (p0 + p1) + (p2 + p3);
            #pragma unroll
            for (int off = 1; off < 16; off <<= 1)
                rs += __shfl_xor(rs, off, 32);

            const float sc = exp2f(mstat[r] - mn);
            lstat[r] = lstat[r] * sc + rs;
            mstat[r] = mn;
            acc0[r] *= sc; acc1[r] *= sc; acc2[r] *= sc; acc3[r] *= sc;

            const int rl = r + 8 * lg;
            plds[wave][rl * 64 + ln]      = (_Float16)p0;
            plds[wave][rl * 64 + 16 + ln] = (_Float16)p1;
            plds[wave][rl * 64 + 32 + ln] = (_Float16)p2;
            plds[wave][rl * 64 + 48 + ln] = (_Float16)p3;
        }
        __syncthreads();

        // ---- reload P as two A-fragments (keys 0..31 and 32..63) ----
        v16h ap0, ap1;
        {
            const _Float16* pr = &plds[wave][ln * 64 + lg * 8];
            const v8h a0 = *(const v8h*)pr;
            const v8h a1 = *(const v8h*)(pr + 16);
            const v8h a2 = *(const v8h*)(pr + 32);
            const v8h a3 = *(const v8h*)(pr + 48);
            #pragma unroll
            for (int t2 = 0; t2 < 8; ++t2) {
                ap0[t2] = a0[t2]; ap0[8 + t2] = a1[t2];
                ap1[t2] = a2[t2]; ap1[8 + t2] = a3[t2];
            }
        }

        // ---- O += P * V  (V^T layout: vt[b][d][key], contiguous in key) ----
        const _Float16* vbase = vt + (size_t)b * HIDDEN * M_TOT + t * 64;
        const _Float16* v0 = vbase + (size_t)(0 * 16 + ln) * M_TOT;
        const _Float16* v1 = vbase + (size_t)(1 * 16 + ln) * M_TOT;
        const _Float16* v2 = vbase + (size_t)(2 * 16 + ln) * M_TOT;
        const _Float16* v3 = vbase + (size_t)(3 * 16 + ln) * M_TOT;
        acc0 = __builtin_amdgcn_wmma_f32_16x16x32_f16(false, ap0, false, load_b16(v0,      lg), (short)0, acc0, false, false);
        acc0 = __builtin_amdgcn_wmma_f32_16x16x32_f16(false, ap1, false, load_b16(v0 + 32, lg), (short)0, acc0, false, false);
        acc1 = __builtin_amdgcn_wmma_f32_16x16x32_f16(false, ap0, false, load_b16(v1,      lg), (short)0, acc1, false, false);
        acc1 = __builtin_amdgcn_wmma_f32_16x16x32_f16(false, ap1, false, load_b16(v1 + 32, lg), (short)0, acc1, false, false);
        acc2 = __builtin_amdgcn_wmma_f32_16x16x32_f16(false, ap0, false, load_b16(v2,      lg), (short)0, acc2, false, false);
        acc2 = __builtin_amdgcn_wmma_f32_16x16x32_f16(false, ap1, false, load_b16(v2 + 32, lg), (short)0, acc2, false, false);
        acc3 = __builtin_amdgcn_wmma_f32_16x16x32_f16(false, ap0, false, load_b16(v3,      lg), (short)0, acc3, false, false);
        acc3 = __builtin_amdgcn_wmma_f32_16x16x32_f16(false, ap1, false, load_b16(v3 + 32, lg), (short)0, acc3, false, false);
    }

    // ---- normalize and write per-head output (b, m, h*64 + d) ----
    #pragma unroll
    for (int r = 0; r < 8; ++r) {
        const int mr = mbase + r + 8 * lg;
        const float inv = 1.f / lstat[r];
        const size_t o = ((size_t)b * M_TOT + mr) * ALLH + h * HIDDEN;
        attn[o + 0 * 16 + ln] = acc0[r] * inv;
        attn[o + 1 * 16 + ln] = acc1[r] * inv;
        attn[o + 2 * 16 + ln] = acc2[r] * inv;
        attn[o + 3 * 16 + ln] = acc3[r] * inv;
    }
}

// ---------------------------------------------------------------------------
// Kernel 3: output projection  out = attn(8192x512) @ Wv(512x64) + bv
// ---------------------------------------------------------------------------
__global__ void __launch_bounds__(256)
out_proj_kernel(const float* __restrict__ attn,
                const float* __restrict__ Wv,
                const float* __restrict__ bv,
                float* __restrict__ out)
{
    const int o   = blockIdx.x * 256 + threadIdx.x;
    const int row = o >> 6;
    const int c   = o & 63;
    const float* ap = attn + (size_t)row * ALLH;
    float s = bv[c];
    #pragma unroll 8
    for (int kk = 0; kk < ALLH; ++kk)
        s += ap[kk] * Wv[kk * HIDDEN + c];
    out[o] = s;
}

// ---------------------------------------------------------------------------
extern "C" void kernel_launch(void* const* d_in, const int* in_sizes, int n_in,
                              void* d_out, int out_size, void* d_ws, size_t ws_size,
                              hipStream_t stream)
{
    const float* hs      = (const float*)d_in[0];
    // d_in[1] = attention_mask (unused by the reference)
    const float* row_emb = (const float*)d_in[2];
    const float* col_emb = (const float*)d_in[3];
    const float* Wq      = (const float*)d_in[4];
    const float* bq      = (const float*)d_in[5];
    const float* Wk      = (const float*)d_in[6];
    const float* bk      = (const float*)d_in[7];
    const float* Wv      = (const float*)d_in[8];
    const float* bv      = (const float*)d_in[9];
    float* out = (float*)d_out;

    char* ws = (char*)d_ws;
    _Float16* qh = (_Float16*)ws;  ws += (size_t)BB * NUM_HEADS * M_TOT * HIDDEN * 2;  // 8 MB
    _Float16* kh = (_Float16*)ws;  ws += (size_t)BB * NUM_HEADS * M_TOT * HIDDEN * 2;  // 8 MB
    _Float16* vt = (_Float16*)ws;  ws += (size_t)BB * HIDDEN * M_TOT * 2;              // 1 MB
    float* brow  = (float*)ws;     ws += (size_t)BB * NUM_HEADS * M_TOT * MAXPOS * 4;  // 8 MB
    float* bcol  = (float*)ws;     ws += (size_t)BB * NUM_HEADS * M_TOT * MAXPOS * 4;  // 8 MB
    float* attn  = (float*)ws;     ws += (size_t)BB * M_TOT * ALLH * 4;                // 16 MB

    proj_kernel<<<BB * M_TOT, 256, 0, stream>>>(hs, row_emb, col_emb,
                                                Wq, bq, Wk, bk,
                                                qh, kh, vt, brow, bcol);
    attn_kernel<<<BB * NUM_HEADS * 16, 128, 0, stream>>>(qh, kh, vt, brow, bcol, attn);
    out_proj_kernel<<<(BB * M_TOT * HIDDEN) / 256, 256, 0, stream>>>(attn, Wv, bv, out);
}
